// NemotronHMoE_10746008175006
// MI455X (gfx1250) — matrix operand order repaired and checked by reading, not compiled, not run on hardware
//
#include <hip/hip_runtime.h>
#include <hip/hip_bf16.h>
#include <stdint.h>

// Problem constants (NemotronH MoE)
#define T_TOK 8192
#define H_DIM 1024
#define E_EXP 512
#define I_DIM 512
#define K_TOP 6
#define G_GRP 8
#define TG_TOP 4
#define CAPQ 1536
#define RSCALE 2.5f

typedef __attribute__((ext_vector_type(16))) __bf16 v16bf;
typedef __attribute__((ext_vector_type(8)))  __bf16 v8bf;
typedef __attribute__((ext_vector_type(8)))  float  v8f;

// round fp32 to bf16 (RNE) keeping result in the high 16 bits of the dword
__device__ __forceinline__ unsigned bfround(float f) {
  unsigned u = __builtin_bit_cast(unsigned, f);
  return u + 0x7FFFu + ((u >> 16) & 1u);
}

// pack two fp32 -> two bf16 with a single v_perm_b32 combine:
// dst = { hi16(rb), hi16(ra) }  (a in low half, b in high half)
__device__ __forceinline__ unsigned bfpack(float a, float b) {
  unsigned ra = bfround(a);
  unsigned rb = bfround(b);
  // sel bytes: dst0=S1.b2, dst1=S1.b3, dst2=S0.b2, dst3=S0.b3  (S1=ra, S0=rb)
  return __builtin_amdgcn_perm(rb, ra, 0x07060302u);
}

__device__ __forceinline__ __bf16 f2bf(float f) {
  unsigned short s = (unsigned short)(bfround(f) >> 16);
  return __builtin_bit_cast(__bf16, s);
}

__device__ __forceinline__ v8f vzero8() {
  v8f z;
#pragma unroll
  for (int i = 0; i < 8; ++i) z[i] = 0.0f;
  return z;
}

// Stage an R x C fp32 tile (global, row stride gld) -> bf16 LDS (row stride C).
// 128-bit global loads + 64-bit LDS stores. C must be a multiple of 4.
template <int R, int C>
__device__ __forceinline__ void stage_tile(const float* __restrict__ g, int gld,
                                           __bf16* __restrict__ s, int tid, int nthr) {
  constexpr int N4 = (R * C) / 4;
  constexpr int C4 = C / 4;
#pragma unroll
  for (int i = tid; i < N4; i += nthr) {
    int r  = i / C4;
    int c4 = i % C4;
    float4 f = *(const float4*)(g + (size_t)r * gld + c4 * 4);
    uint2 u2;
    u2.x = bfpack(f.x, f.y);
    u2.y = bfpack(f.z, f.w);
    *(uint2*)(s + i * 4) = u2;
  }
}

// A fragment: 16xK row-major bf16 tile in LDS, stride lda, K-window at k0.
// CDNA5 layout: lane = m + 16*kh; elems 0..7 -> K = kh*8+e ; elems 8..15 -> K = 16+kh*8+(e-8)
__device__ __forceinline__ v16bf load_frag_a(const __bf16* A, int lda, int k0, int lane) {
  int m  = lane & 15;
  int kh = (lane >> 4) & 1;
  const __bf16* p = A + m * lda + k0 + kh * 8;
  union { v16bf v; v8bf h[2]; } u;
  u.h[0] = *(const v8bf*)(p);
  u.h[1] = *(const v8bf*)(p + 16);
  return u.v;
}

// B fragment: 32x16 from row-major KxN bf16 tile in LDS (stride ldb), col offset n0.
// CDNA5 layout: lane = n + 16*kh; elems 0..7 -> K = kh*8+e ; elems 8..15 -> K = 16+kh*8+(e-8)
__device__ __forceinline__ v16bf load_frag_b(const __bf16* B, int ldb, int n0, int lane) {
  int n  = lane & 15;
  int kh = (lane >> 4) & 1;
  const __bf16* p = B + (kh * 8) * ldb + n0 + n;
  v16bf v;
#pragma unroll
  for (int e = 0; e < 8; ++e) {
    v[e]     = p[e * ldb];
    v[8 + e] = p[(16 + e) * ldb];
  }
  return v;
}

#define WMMA_BF16(a, b, c) \
  __builtin_amdgcn_wmma_f32_16x16x32_bf16(false, (a), false, (b), (short)0, (c), false, false)

// ---------------------------------------------------------------------------
// K1: scores[T,E] = sigmoid(x @ gate_w)   (bf16 WMMA, 32x64 tile per block)
// ---------------------------------------------------------------------------
__global__ __launch_bounds__(256) void router_kernel(
    const float* __restrict__ x, const float* __restrict__ gate_w,
    float* __restrict__ scores) {
  __shared__ __align__(16) __bf16 As[32 * 32];
  __shared__ __align__(16) __bf16 Bs[32 * 64];
  const int tid = threadIdx.x, lane = tid & 31, wid = tid >> 5;
  const int mwave = wid >> 2;   // 0..1
  const int nwave = wid & 3;    // 0..3
  const int n_base = blockIdx.x * 64;
  const int m_base = blockIdx.y * 32;
  v8f acc = vzero8();
  for (int k0 = 0; k0 < H_DIM; k0 += 32) {
    stage_tile<32, 32>(x + (size_t)m_base * H_DIM + k0, H_DIM, As, tid, 256);
    stage_tile<32, 64>(gate_w + (size_t)k0 * E_EXP + n_base, E_EXP, Bs, tid, 256);
    __syncthreads();
    v16bf a = load_frag_a(As + mwave * 16 * 32, 32, 0, lane);
    v16bf b = load_frag_b(Bs, 64, nwave * 16, lane);
    acc = WMMA_BF16(a, b, acc);
    __syncthreads();
  }
  const int n  = n_base + nwave * 16 + (lane & 15);
  const int mb = m_base + mwave * 16 + ((lane >> 4) << 3);
#pragma unroll
  for (int v = 0; v < 8; ++v) {
    float s = 1.0f / (1.0f + __expf(-acc[v]));
    scores[(size_t)(mb + v) * E_EXP + n] = s;
  }
}

// ---------------------------------------------------------------------------
// K2: zero per-expert counters
// ---------------------------------------------------------------------------
__global__ void zero_counts_kernel(int* __restrict__ counts) {
  int i = blockIdx.x * blockDim.x + threadIdx.x;
  if (i < E_EXP) counts[i] = 0;
}

// ---------------------------------------------------------------------------
// K3: grouped sigmoid top-k routing + capacity dispatch (1 thread / token)
// ---------------------------------------------------------------------------
__global__ __launch_bounds__(256) void route_topk_kernel(
    const float* __restrict__ scores, const float* __restrict__ bias,
    float* __restrict__ topk_w, int* __restrict__ counts,
    int* __restrict__ expert_tok) {
  int t = blockIdx.x * blockDim.x + threadIdx.x;
  if (t >= T_TOK) return;
  const float* srow = scores + (size_t)t * E_EXP;
  // group score = sum of top-2 (score + bias) per group of 64
  float gscore[G_GRP];
#pragma unroll
  for (int g = 0; g < G_GRP; ++g) {
    float m1 = -3.4e38f, m2 = -3.4e38f;
    const int base = g * (E_EXP / G_GRP);
    for (int j = 0; j < E_EXP / G_GRP; ++j) {
      float v = srow[base + j] + bias[base + j];
      if (v > m1) { m2 = m1; m1 = v; }
      else if (v > m2) { m2 = v; }
    }
    gscore[g] = m1 + m2;
  }
  // top-4 groups
  unsigned gmask = 0;
#pragma unroll
  for (int r = 0; r < TG_TOP; ++r) {
    int best = 0; float bv = -3.4e38f;
    for (int g = 0; g < G_GRP; ++g)
      if (!((gmask >> g) & 1u) && gscore[g] > bv) { bv = gscore[g]; best = g; }
    gmask |= 1u << best;
  }
  // top-6 experts among allowed groups (insertion sort, earlier index wins ties)
  int ids[K_TOP]; float vals[K_TOP];
#pragma unroll
  for (int k = 0; k < K_TOP; ++k) { ids[k] = 0; vals[k] = -3.4e38f; }
  for (int e = 0; e < E_EXP; ++e) {
    if (!((gmask >> (e >> 6)) & 1u)) continue;
    float v = srow[e] + bias[e];
    if (v > vals[K_TOP - 1]) {
      int p = K_TOP - 1;
      while (p > 0 && v > vals[p - 1]) {
        vals[p] = vals[p - 1]; ids[p] = ids[p - 1]; --p;
      }
      vals[p] = v; ids[p] = e;
    }
  }
  // weights from raw sigmoid scores, normalized, pre-scaled
  float w[K_TOP]; float wsum = 0.f;
#pragma unroll
  for (int k = 0; k < K_TOP; ++k) { w[k] = srow[ids[k]]; wsum += w[k]; }
  float inv = RSCALE / wsum;
#pragma unroll
  for (int k = 0; k < K_TOP; ++k) {
    topk_w[t * K_TOP + k] = w[k] * inv;
    int e = ids[k];
    int pos = atomicAdd(&counts[e], 1);
    if (pos < CAPQ) expert_tok[e * CAPQ + pos] = (t << 3) | k;
  }
}

// ---------------------------------------------------------------------------
// K4: shared expert  out = relu2(x @ up) @ down   (writes d_out)
// ---------------------------------------------------------------------------
__global__ __launch_bounds__(256) void shared_ffn_kernel(
    const float* __restrict__ x, const float* __restrict__ up,
    const float* __restrict__ down, float* __restrict__ out) {
  __shared__ __align__(16) __bf16 As[16 * 32];       //  1 KB
  __shared__ __align__(16) __bf16 Ss[16 * H_DIM];    // 32 KB
  __shared__ __align__(16) __bf16 Bs[32 * 256];      // 16 KB
  const int tid = threadIdx.x, lane = tid & 31, wid = tid >> 5;
  const int m_base = blockIdx.x * 16;
  const int mrow = (lane >> 4) << 3;
  // Phase 1: S = relu2(x @ up), N = 1024 in 4 passes of 256
  for (int p = 0; p < H_DIM / 256; ++p) {
    v8f acc[2]; acc[0] = vzero8(); acc[1] = vzero8();
    const int pn = p * 256;
    for (int k0 = 0; k0 < H_DIM; k0 += 32) {
      stage_tile<16, 32>(x + (size_t)m_base * H_DIM + k0, H_DIM, As, tid, 256);
      stage_tile<32, 256>(up + (size_t)k0 * H_DIM + pn, H_DIM, Bs, tid, 256);
      __syncthreads();
      v16bf a = load_frag_a(As, 32, 0, lane);
#pragma unroll
      for (int j = 0; j < 2; ++j) {
        v16bf b = load_frag_b(Bs, 256, wid * 32 + j * 16, lane);
        acc[j] = WMMA_BF16(a, b, acc[j]);
      }
      __syncthreads();
    }
#pragma unroll
    for (int j = 0; j < 2; ++j) {
      int n = pn + wid * 32 + j * 16 + (lane & 15);
#pragma unroll
      for (int v = 0; v < 8; ++v) {
        float val = acc[j][v];
        val = val > 0.f ? val * val : 0.f;
        Ss[(mrow + v) * H_DIM + n] = f2bf(val);
      }
    }
    __syncthreads();
  }
  // Phase 2: out = S @ down
  for (int p = 0; p < H_DIM / 256; ++p) {
    v8f acc[2]; acc[0] = vzero8(); acc[1] = vzero8();
    const int pn = p * 256;
    for (int k0 = 0; k0 < H_DIM; k0 += 32) {
      stage_tile<32, 256>(down + (size_t)k0 * H_DIM + pn, H_DIM, Bs, tid, 256);
      __syncthreads();
      v16bf a = load_frag_a(Ss, H_DIM, k0, lane);
#pragma unroll
      for (int j = 0; j < 2; ++j) {
        v16bf b = load_frag_b(Bs, 256, wid * 32 + j * 16, lane);
        acc[j] = WMMA_BF16(a, b, acc[j]);
      }
      __syncthreads();
    }
#pragma unroll
    for (int j = 0; j < 2; ++j) {
      int n = pn + wid * 32 + j * 16 + (lane & 15);
#pragma unroll
      for (int v = 0; v < 8; ++v)
        out[(size_t)(m_base + mrow + v) * H_DIM + n] = acc[j][v];
    }
    __syncthreads();
  }
}

// ---------------------------------------------------------------------------
// K5: routed experts, grouped GEMM over gathered tokens; atomic combine.
// grid = (CAPQ/16, E); blocks past the expert's count exit (uniform).
// ---------------------------------------------------------------------------
__global__ __launch_bounds__(256) void expert_ffn_kernel(
    const float* __restrict__ x, const float* __restrict__ w_up,
    const float* __restrict__ w_down, const int* __restrict__ counts,
    const int* __restrict__ expert_tok, const float* __restrict__ topk_w,
    float* __restrict__ out) {
  const int e = blockIdx.y;
  const int tile = blockIdx.x;
  int cnt = counts[e]; if (cnt > CAPQ) cnt = CAPQ;
  if (tile * 16 >= cnt) return;   // block-uniform: EXEC stays all-1 for WMMA
  __shared__ __align__(16) __bf16 As[16 * 32];       //  1 KB
  __shared__ __align__(16) __bf16 Hs[16 * I_DIM];    // 16 KB
  __shared__ __align__(16) __bf16 Bs[32 * I_DIM];    // 32 KB
  __shared__ int   s_tok[16];
  __shared__ float s_w[16];
  const int tid = threadIdx.x, lane = tid & 31, wid = tid >> 5;
  const int mrow = (lane >> 4) << 3;
  if (tid < 16) {
    int idx = tile * 16 + tid;
    if (idx < cnt) {
      int raw = expert_tok[e * CAPQ + idx];
      s_tok[tid] = raw >> 3;
      s_w[tid]   = topk_w[(raw >> 3) * K_TOP + (raw & 7)];
    } else { s_tok[tid] = -1; s_w[tid] = 0.f; }
  }
  __syncthreads();
  const float* wu = w_up   + (size_t)e * H_DIM * I_DIM;
  const float* wd = w_down + (size_t)e * I_DIM * H_DIM;
  // Up: H[16,512] = gather(x)[16,1024] @ wu[1024,512]
  v8f acc[4];
#pragma unroll
  for (int j = 0; j < 4; ++j) acc[j] = vzero8();
  for (int k0 = 0; k0 < H_DIM; k0 += 32) {
    // gathered A tile: per-row token lookup, float4 loads, b64 LDS stores
#pragma unroll
    for (int i = tid; i < (16 * 32) / 4; i += 256) {
      int r = i >> 3, c4 = i & 7;
      int tk = s_tok[r];
      uint2 u2;
      if (tk >= 0) {
        float4 f = *(const float4*)(x + (size_t)tk * H_DIM + k0 + c4 * 4);
        u2.x = bfpack(f.x, f.y);
        u2.y = bfpack(f.z, f.w);
      } else { u2.x = 0u; u2.y = 0u; }
      *(uint2*)(As + i * 4) = u2;
    }
    stage_tile<32, 512>(wu + (size_t)k0 * I_DIM, I_DIM, Bs, tid, 256);
    __syncthreads();
    v16bf a = load_frag_a(As, 32, 0, lane);
#pragma unroll
    for (int j = 0; j < 4; ++j) {
      v16bf b = load_frag_b(Bs, I_DIM, wid * 64 + j * 16, lane);
      acc[j] = WMMA_BF16(a, b, acc[j]);
    }
    __syncthreads();
  }
#pragma unroll
  for (int j = 0; j < 4; ++j) {
    int n = wid * 64 + j * 16 + (lane & 15);
#pragma unroll
    for (int v = 0; v < 8; ++v) {
      float val = acc[j][v];
      val = val > 0.f ? val * val : 0.f;
      Hs[(mrow + v) * I_DIM + n] = f2bf(val);
    }
  }
  __syncthreads();
  // Down: Y[16,1024] = H[16,512] @ wd[512,1024], two passes of N=512
  for (int p = 0; p < 2; ++p) {
    v8f acc2[4];
#pragma unroll
    for (int j = 0; j < 4; ++j) acc2[j] = vzero8();
    const int pn = p * 512;
    for (int k0 = 0; k0 < I_DIM; k0 += 32) {
      stage_tile<32, 512>(wd + (size_t)k0 * H_DIM + pn, H_DIM, Bs, tid, 256);
      __syncthreads();
      v16bf a = load_frag_a(Hs, I_DIM, k0, lane);
#pragma unroll
      for (int j = 0; j < 4; ++j) {
        v16bf b = load_frag_b(Bs, 512, wid * 64 + j * 16, lane);
        acc2[j] = WMMA_BF16(a, b, acc2[j]);
      }
      __syncthreads();
    }
#pragma unroll
    for (int j = 0; j < 4; ++j) {
      int n = pn + wid * 64 + j * 16 + (lane & 15);
#pragma unroll
      for (int v = 0; v < 8; ++v) {
        int m = mrow + v;
        int tk = s_tok[m];
        if (tk >= 0)
          atomicAdd(&out[(size_t)tk * H_DIM + n], acc2[j][v] * s_w[m]);
      }
    }
    __syncthreads();
  }
}

// ---------------------------------------------------------------------------
extern "C" void kernel_launch(void* const* d_in, const int* in_sizes, int n_in,
                              void* d_out, int out_size, void* d_ws, size_t ws_size,
                              hipStream_t stream) {
  const float* x      = (const float*)d_in[0];
  const float* gate_w = (const float*)d_in[1];
  const float* gate_b = (const float*)d_in[2];
  const float* w_up   = (const float*)d_in[3];
  const float* w_down = (const float*)d_in[4];
  const float* sh_up  = (const float*)d_in[5];
  const float* sh_dn  = (const float*)d_in[6];
  float* out = (float*)d_out;

  char* ws = (char*)d_ws;
  size_t off = 0;
  float* scores = (float*)(ws + off); off += (size_t)T_TOK * E_EXP * sizeof(float);
  float* topk_w = (float*)(ws + off); off += (size_t)T_TOK * K_TOP * sizeof(float);
  int*   counts = (int*)(ws + off);   off += (size_t)E_EXP * sizeof(int);
  off = (off + 255) & ~(size_t)255;
  int* expert_tok = (int*)(ws + off); off += (size_t)E_EXP * CAPQ * sizeof(int);
  (void)ws_size; (void)in_sizes; (void)n_in; (void)out_size;

  router_kernel<<<dim3(E_EXP / 64, T_TOK / 32), 256, 0, stream>>>(x, gate_w, scores);
  zero_counts_kernel<<<2, 256, 0, stream>>>(counts);
  route_topk_kernel<<<T_TOK / 256, 256, 0, stream>>>(scores, gate_b, topk_w, counts, expert_tok);
  shared_ffn_kernel<<<T_TOK / 16, 256, 0, stream>>>(x, sh_up, sh_dn, out);
  expert_ffn_kernel<<<dim3(CAPQ / 16, E_EXP), 256, 0, stream>>>(
      x, w_up, w_down, counts, expert_tok, topk_w, out);
}
